// Head_12996571037869
// MI455X (gfx1250) — matrix operand order repaired and checked by reading, not compiled, not run on hardware
//
#include <hip/hip_runtime.h>

typedef unsigned int u32;
typedef __attribute__((ext_vector_type(16))) _Float16 v16h;
typedef __attribute__((ext_vector_type(8)))  float    v8f;

union Frag { v16h h; u32 u[8]; };

__device__ __forceinline__ u32 pack2h(float a, float b) {
  union { _Float16 h[2]; u32 u; } p;
  p.h[0] = (_Float16)a; p.h[1] = (_Float16)b;
  return p.u;
}

#define SEQ 4096
#define EMB 1024
#define HD  128
#define NB  4
#define ROWS (NB*SEQ)   // 16384

// ---- optional CDNA5 async global->LDS copy path --------------------------
#if defined(__has_builtin)
#if __has_builtin(__builtin_amdgcn_global_load_async_to_lds_b128) && \
    __has_builtin(__builtin_amdgcn_s_wait_asynccnt)
#define USE_ASYNC 1
#endif
#endif
#ifndef USE_ASYNC
#define USE_ASYNC 0
#endif

#if USE_ASYNC
typedef __attribute__((__vector_size__(16))) int v4i;
typedef __attribute__((address_space(1))) v4i g_v4i;
typedef __attribute__((address_space(3))) v4i l_v4i;
__device__ __forceinline__ void async_cp16(const u32* g, u32* l) {
  __builtin_amdgcn_global_load_async_to_lds_b128(
      (g_v4i*)g, (l_v4i*)l, 0, 0);
}
#endif

// ---------------------------------------------------------------------------
// Kernel 1: one projection  Y = X @ W^T.  X:[16384,1024] f32, W:[128,1024] f32.
// MODE 0/1: row-major f16 out [16384,128].  MODE 2: transposed out [B][H][S].
// Block = 256 thr (8 waves) -> 128x128 output tile, K staged 32 at a time.
// ---------------------------------------------------------------------------
template <int MODE>
__global__ __launch_bounds__(256) void proj_kernel(
    const float* __restrict__ x, const float* __restrict__ W,
    _Float16* __restrict__ out)
{
  __shared__ __align__(16) u32 lds_x[128][16];   // 128 x 32 f16
  __shared__ __align__(16) u32 lds_w[128][16];

  const int t   = threadIdx.x;
  const int wvi = __builtin_amdgcn_readfirstlane(t >> 5);
  const int L   = t & 31;
  const int hl  = L >> 4;
  const int l16 = L & 15;
  const int rbase = blockIdx.x * 128;

  v8f acc[8];
  for (int n = 0; n < 8; ++n)
    for (int r = 0; r < 8; ++r) acc[n][r] = 0.f;

  for (int kb = 0; kb < EMB; kb += 32) {
    for (int f = 0; f < 4; ++f) {
      int lin = t + f * 256;          // float4 index
      int row = lin >> 3;
      int c4  = lin & 7;
      float4 xv = *(const float4*)(x + (size_t)(rbase + row) * EMB + kb + c4 * 4);
      lds_x[row][c4 * 2]     = pack2h(xv.x, xv.y);
      lds_x[row][c4 * 2 + 1] = pack2h(xv.z, xv.w);
      float4 wv4 = *(const float4*)(W + (size_t)row * EMB + kb + c4 * 4);
      lds_w[row][c4 * 2]     = pack2h(wv4.x, wv4.y);
      lds_w[row][c4 * 2 + 1] = pack2h(wv4.z, wv4.w);
    }
    __syncthreads();

    Frag a;                            // A: 16x32 f16 (this wave's rows)
    for (int v = 0; v < 8; ++v)
      a.u[v] = lds_x[wvi * 16 + l16][hl * 8 + v];

    Frag bf[2];                        // ping-pong B fragments
    for (int v = 0; v < 8; ++v) bf[0].u[v] = lds_w[l16][hl * 8 + v];
#pragma unroll
    for (int n = 0; n < 8; ++n) {
      if (n < 7)
        for (int v = 0; v < 8; ++v)
          bf[(n + 1) & 1].u[v] = lds_w[(n + 1) * 16 + l16][hl * 8 + v];
      acc[n] = __builtin_amdgcn_wmma_f32_16x16x32_f16(
                 false, a.h, false, bf[n & 1].h, (short)0, acc[n], false, false);
    }
    __syncthreads();
  }

  const int growb = rbase + wvi * 16;
  if (MODE == 2) {
    // lane owns 8 consecutive s positions per N-tile -> one b128 store
    const int slb  = growb + hl * 8;   // stays inside one batch (mult of 8)
    const int bb   = slb >> 12;
    const int sl   = slb & (SEQ - 1);
#pragma unroll
    for (int n = 0; n < 8; ++n) {
      int h = n * 16 + l16;
      uint4 w4;
      w4.x = pack2h(acc[n][0], acc[n][1]);
      w4.y = pack2h(acc[n][2], acc[n][3]);
      w4.z = pack2h(acc[n][4], acc[n][5]);
      w4.w = pack2h(acc[n][6], acc[n][7]);
      *(uint4*)&out[(((size_t)bb * HD + h) << 12) + sl] = w4;
    }
  } else {
#pragma unroll
    for (int n = 0; n < 8; ++n)
      for (int r = 0; r < 8; ++r) {
        int grow = growb + hl * 8 + r;
        out[(size_t)grow * HD + n * 16 + l16] = (_Float16)acc[n][r];
      }
  }
}

// ---------------------------------------------------------------------------
// Kernel 2: causal flash attention.  128 query rows/block (8 waves x 16),
// 32-key tiles, double-buffered K/V staging (async-to-LDS when available).
// scale = EMB^-0.5 = 1/32 (reference scales by embedding size).
// ---------------------------------------------------------------------------
__global__ __launch_bounds__(256) void attn_kernel(
    const _Float16* __restrict__ Qh, const _Float16* __restrict__ Kh,
    const _Float16* __restrict__ Vt, float* __restrict__ out)
{
  __shared__ __align__(16) u32 lds_k[2][32][64];    // 32 keys x 128 h   16KB
  __shared__ __align__(16) u32 lds_v[2][128][16];   // 128 h  x 32 keys  16KB
  __shared__ __align__(16) u32 lds_p[8][16][16];    // per-wave P tile    8KB

  const int t   = threadIdx.x;
  const int wvi = __builtin_amdgcn_readfirstlane(t >> 5);
  const int L   = t & 31;
  const int hl  = L >> 4;
  const int l16 = L & 15;
  const int qt  = blockIdx.x;
  const int b   = blockIdx.y;
  const int wq  = qt * 128 + wvi * 16;
  const float scale = 0.03125f;

  const u32* Q32 = (const u32*)Qh;   // [16384][64]
  const u32* K32 = (const u32*)Kh;
  const u32* V32 = (const u32*)Vt;   // [NB][128][2048]

  // cooperative stage of one 32-key K tile + V^T tile into buffer `buf`
  auto stage_kv = [&](int j, int buf) {
    const int kb = j * 32;
#pragma unroll
    for (int f = 0; f < 2; ++f) {              // K: 512 x 16B chunks
      int lin = t + f * 256;
      int row = lin >> 4, col = (lin & 15) * 4;
      const u32* g = &K32[(size_t)(b * SEQ + kb + row) * 64 + col];
      u32* l = &lds_k[buf][row][col];
#if USE_ASYNC
      async_cp16(g, l);
#else
      *(uint4*)l = *(const uint4*)g;
#endif
    }
#pragma unroll
    for (int f = 0; f < 2; ++f) {              // V^T: 512 x 16B chunks
      int lin = t + f * 256;
      int row = lin >> 2, col = (lin & 3) * 4;
      const u32* g = &V32[((size_t)b * HD + row) * 2048 + (kb >> 1) + col];
      u32* l = &lds_v[buf][row][col];
#if USE_ASYNC
      async_cp16(g, l);
#else
      *(uint4*)l = *(const uint4*)g;
#endif
    }
  };

  // Q fragments for this wave's 16 rows: 4 chunks of 32 over h
  Frag qf[4];
  {
    size_t qrow = (size_t)(b * SEQ + wq + l16) * 64;
    for (int c = 0; c < 4; ++c)
      for (int v = 0; v < 8; ++v)
        qf[c].u[v] = Q32[qrow + c * 16 + hl * 8 + v];
  }

  v8f o[8];
  float mrow[8], lrow[8];
  for (int n = 0; n < 8; ++n)
    for (int r = 0; r < 8; ++r) o[n][r] = 0.f;
  for (int r = 0; r < 8; ++r) { mrow[r] = -3e30f; lrow[r] = 0.f; }

  const int jmax = 4 * qt + 4;
  stage_kv(0, 0);
#if USE_ASYNC
  __builtin_amdgcn_s_wait_asynccnt(0);
#endif
  __syncthreads();

  for (int j = 0; j < jmax; ++j) {
    const int kb  = j * 32;
    const int cur = j & 1;

    if (j + 1 < jmax) stage_kv(j + 1, cur ^ 1);   // overlaps compute below

    if (kb <= wq + 15) {                          // wave-uniform causal skip
      // ---- S = Q K^T (16x32), ping-pong B fragments ----
      v8f s0, s1;
      for (int r = 0; r < 8; ++r) { s0[r] = 0.f; s1[r] = 0.f; }
      Frag B0[2], B1[2];
      for (int v = 0; v < 8; ++v) {
        B0[0].u[v] = lds_k[cur][l16][hl * 8 + v];
        B1[0].u[v] = lds_k[cur][16 + l16][hl * 8 + v];
      }
#pragma unroll
      for (int c = 0; c < 4; ++c) {
        if (c < 3)
          for (int v = 0; v < 8; ++v) {
            B0[(c + 1) & 1].u[v] = lds_k[cur][l16][(c + 1) * 16 + hl * 8 + v];
            B1[(c + 1) & 1].u[v] = lds_k[cur][16 + l16][(c + 1) * 16 + hl * 8 + v];
          }
        s0 = __builtin_amdgcn_wmma_f32_16x16x32_f16(
               false, qf[c].h, false, B0[c & 1].h, (short)0, s0, false, false);
        s1 = __builtin_amdgcn_wmma_f32_16x16x32_f16(
               false, qf[c].h, false, B1[c & 1].h, (short)0, s1, false, false);
      }

      // ---- online softmax ----
      float al[8];
#pragma unroll
      for (int r = 0; r < 8; ++r) {
        int qrow = wq + hl * 8 + r;
        float a0 = (kb + l16      <= qrow) ? s0[r] * scale : -3e30f;
        float a1 = (kb + 16 + l16 <= qrow) ? s1[r] * scale : -3e30f;
        float mx = fmaxf(a0, a1);
        mx = fmaxf(mx, __shfl_xor(mx, 1, 32));
        mx = fmaxf(mx, __shfl_xor(mx, 2, 32));
        mx = fmaxf(mx, __shfl_xor(mx, 4, 32));
        mx = fmaxf(mx, __shfl_xor(mx, 8, 32));
        float mnew = fmaxf(mrow[r], mx);
        al[r] = __expf(mrow[r] - mnew);
        float p0 = __expf(a0 - mnew);
        float p1 = __expf(a1 - mnew);
        s0[r] = p0; s1[r] = p1;
        float ls = p0 + p1;
        ls += __shfl_xor(ls, 1, 32);
        ls += __shfl_xor(ls, 2, 32);
        ls += __shfl_xor(ls, 4, 32);
        ls += __shfl_xor(ls, 8, 32);
        lrow[r] = al[r] * lrow[r] + ls;
        mrow[r] = mnew;
      }
      for (int n = 0; n < 8; ++n)
        for (int r = 0; r < 8; ++r) o[n][r] *= al[r];

      // ---- P: C-layout -> A-fragment via wave-private LDS ----
      _Float16* pp = (_Float16*)&lds_p[wvi][0][0];
#pragma unroll
      for (int r = 0; r < 8; ++r) {
        int m = hl * 8 + r;
        pp[m * 32 + l16]      = (_Float16)s0[r];
        pp[m * 32 + 16 + l16] = (_Float16)s1[r];
      }
      asm volatile("" ::: "memory");   // same-wave DS ops execute in order
      Frag pf;
      for (int v = 0; v < 8; ++v)
        pf.u[v] = lds_p[wvi][l16][hl * 8 + v];

      // ---- O += P V, ping-pong V fragments ----
      Frag vf[2];
      for (int v = 0; v < 8; ++v)
        vf[0].u[v] = lds_v[cur][l16][hl * 8 + v];
#pragma unroll
      for (int n = 0; n < 8; ++n) {
        if (n < 7)
          for (int v = 0; v < 8; ++v)
            vf[(n + 1) & 1].u[v] = lds_v[cur][(n + 1) * 16 + l16][hl * 8 + v];
        o[n] = __builtin_amdgcn_wmma_f32_16x16x32_f16(
                 false, pf.h, false, vf[n & 1].h, (short)0, o[n], false, false);
      }
    }
#if USE_ASYNC
    __builtin_amdgcn_s_wait_asynccnt(0);
#endif
    __syncthreads();
  }

  // ---- epilogue ----
  float inv[8];
  for (int r = 0; r < 8; ++r) inv[r] = 1.f / lrow[r];
  size_t obase = ((size_t)b * SEQ + wq) * HD;
#pragma unroll
  for (int n = 0; n < 8; ++n)
    for (int r = 0; r < 8; ++r) {
      int m = hl * 8 + r;
      out[obase + (size_t)m * HD + n * 16 + l16] = o[n][r] * inv[r];
    }
}

// ---------------------------------------------------------------------------
extern "C" void kernel_launch(void* const* d_in, const int* in_sizes, int n_in,
                              void* d_out, int out_size, void* d_ws, size_t ws_size,
                              hipStream_t stream) {
  const float* x  = (const float*)d_in[0];
  const float* Wk = (const float*)d_in[1];
  const float* Wq = (const float*)d_in[2];
  const float* Wv = (const float*)d_in[3];
  float* out = (float*)d_out;

  char* ws = (char*)d_ws;
  _Float16* Qh = (_Float16*)(ws);                                // 4 MB
  _Float16* Kh = (_Float16*)(ws + (size_t)ROWS * HD * 2);        // 4 MB
  _Float16* Vt = (_Float16*)(ws + (size_t)2 * ROWS * HD * 2);    // 4 MB

  proj_kernel<0><<<dim3(ROWS / 128), 256, 0, stream>>>(x, Wq, Qh);
  proj_kernel<1><<<dim3(ROWS / 128), 256, 0, stream>>>(x, Wk, Kh);
  proj_kernel<2><<<dim3(ROWS / 128), 256, 0, stream>>>(x, Wv, Vt);
  attn_kernel<<<dim3(SEQ / 128, NB), 256, 0, stream>>>(Qh, Kh, Vt, out);
}
